// GRU_23587960389856
// MI455X (gfx1250) — 16-bit and 8-bit matrix operand order repaired and checked by reading, compiled, not run on hardware
//
#include <hip/hip_runtime.h>


typedef __attribute__((ext_vector_type(16))) __bf16 v16bf;
typedef __attribute__((ext_vector_type(8)))  float  v8f;
typedef __attribute__((ext_vector_type(8)))  int    v8i;

#define T_STEPS 512
#define IN_F    63
#define HID     32
#define BATCH   4096

__device__ inline unsigned f32_to_e4m3(float x) {
    unsigned u = __float_as_uint(x);
    unsigned s = (u >> 24) & 0x80u;
    float a = fabsf(x);
    if (!(a == a)) return s | 0x7Fu;
    if (a >= 448.0f) return s | 0x7Eu;
    if (a < 0.0009765625f) return s;
    if (a < 0.015625f) {
        int m = (int)(a * 512.0f + 0.5f);
        return s | (unsigned)((m > 7) ? 8 : m);
    }
    int ex; float fr = frexpf(a, &ex);
    int m = (int)(fr * 16.0f + 0.5f);
    if (m == 16) { m = 8; ++ex; }
    int E = ex - 1 + 7;
    if (E >= 16) return s | 0x7Eu;
    return s | (unsigned)(E << 3) | (unsigned)(m & 7);
}

__device__ inline int pack4_fp8(float a, float b, float c, float d) {
#if __has_builtin(__builtin_amdgcn_cvt_pk_fp8_f32)
    int t = __builtin_amdgcn_cvt_pk_fp8_f32(a, b, 0, false);
    t     = __builtin_amdgcn_cvt_pk_fp8_f32(c, d, t, true);
    return t;
#else
    return (int)(f32_to_e4m3(a) | (f32_to_e4m3(b) << 8) |
                 (f32_to_e4m3(c) << 16) | (f32_to_e4m3(d) << 24));
#endif
}

__device__ inline int pack_bf16(float a, float b) {
    unsigned short lo = __builtin_bit_cast(unsigned short, (__bf16)a);
    unsigned short hi = __builtin_bit_cast(unsigned short, (__bf16)b);
    return (int)((unsigned)lo | ((unsigned)hi << 16));
}

__device__ inline v16bf as_v16bf(v8i x) {
    union { v8i i; v16bf b; } u; u.i = x; return u.b;
}

__device__ inline float fast_tanh(float x) {
#if __has_builtin(__builtin_amdgcn_tanhf)
    return __builtin_amdgcn_tanhf(x);
#else
    float e = __expf(-2.0f * x);
    return 2.0f * __builtin_amdgcn_rcpf(1.0f + e) - 1.0f;
#endif
}
__device__ inline float fast_sigmoid(float x) {
#if __has_builtin(__builtin_amdgcn_tanhf)
    return 0.5f + 0.5f * fast_tanh(0.5f * x);
#else
    float e = __expf(-x);
    return __builtin_amdgcn_rcpf(1.0f + e);
#endif
}

__device__ inline v8f wmma_fp8(v8i a, v8i b, v8f c) {
    return __builtin_amdgcn_wmma_f32_16x16x64_fp8_fp8(a, b, (short)0, c, false, false);
}
__device__ inline v8f wmma_bf16(v16bf a, v16bf b, v8f c) {
    return __builtin_amdgcn_wmma_f32_16x16x32_bf16(false, a, false, b,
                                                   (short)0, c, false, false);
}

__global__ __launch_bounds__(32)
void gru_fused_kernel(const float* __restrict__ input,
                      const float* __restrict__ w_xz, const float* __restrict__ w_hz, const float* __restrict__ b_z,
                      const float* __restrict__ w_xr, const float* __restrict__ w_hr, const float* __restrict__ b_r,
                      const float* __restrict__ w_xh, const float* __restrict__ w_hh, const float* __restrict__ b_h,
                      const float* __restrict__ w_out, const float* __restrict__ b_out,
                      float* __restrict__ out)
{
    const int lane = threadIdx.x;
    const int n    = lane & 15;
    const int hh   = lane >> 4;
    const int hi   = hh * 8;
    const int b0   = blockIdx.x * 16;
    const int row  = b0 + n;

    const float* wxp[3] = { w_xz, w_xr, w_xh };
    const float* whp[3] = { w_hz, w_hr, w_hh };
    const float* bbp[3] = { b_z,  b_r,  b_h  };

    v8i   axw[3][2];
    v16bf ahw[3][2];

    #pragma unroll
    for (int g = 0; g < 3; ++g) {
        #pragma unroll
        for (int mt = 0; mt < 2; ++mt) {
            const int mcol = mt * 16 + n;
            v8i a;
            #pragma unroll
            for (int v = 0; v < 8; ++v) {
                float w4[4];
                #pragma unroll
                for (int b = 0; b < 4; ++b) {
                    const int k = (v >> 2) * 32 + 16 * hh + (v & 3) * 4 + b;
                    w4[b] = (k < IN_F) ? wxp[g][k * HID + mcol]
                                       : bbp[g][mcol];
                }
                a[v] = pack4_fp8(w4[0], w4[1], w4[2], w4[3]);
            }
            axw[g][mt] = a;
            v16bf ar;
            #pragma unroll
            for (int e = 0; e < 16; ++e) {
                const int kl = 16 * hh + e;
                ar[e] = (__bf16)whp[g][kl * HID + mcol];
            }
            ahw[g][mt] = ar;
        }
    }

    v8f hst[2], zc;
    #pragma unroll
    for (int j = 0; j < 8; ++j) { hst[0][j] = 0.0f; hst[1][j] = 0.0f; zc[j] = 0.0f; }

    const float* rowbase = input + (size_t)row * (T_STEPS * IN_F);
    const int fb = 16 * hh;

    float xf[32];
    {
        const float* rp = rowbase;
        #pragma unroll
        for (int i = 0; i < 16; ++i) xf[i] = rp[fb + i];
        #pragma unroll
        for (int i = 0; i < 16; ++i) {
            const int f = fb + 32 + i;
            xf[16 + i] = (f < IN_F) ? rp[f] : 1.0f;
        }
    }

    for (int t = 0; t < T_STEPS; ++t) {
        v8i xb;
        #pragma unroll
        for (int d = 0; d < 4; ++d)
            xb[d] = pack4_fp8(xf[4*d], xf[4*d+1], xf[4*d+2], xf[4*d+3]);
        #pragma unroll
        for (int d = 0; d < 4; ++d)
            xb[4+d] = pack4_fp8(xf[16+4*d], xf[16+4*d+1], xf[16+4*d+2], xf[16+4*d+3]);

        if (t + 1 < T_STEPS) {
            const float* rp = rowbase + (size_t)(t + 1) * IN_F;
            #pragma unroll
            for (int i = 0; i < 16; ++i) xf[i] = rp[fb + i];
            #pragma unroll
            for (int i = 0; i < 16; ++i) {
                const int f = fb + 32 + i;
                xf[16 + i] = (f < IN_F) ? rp[f] : 1.0f;
            }
            if (t + 5 < T_STEPS)
                __builtin_prefetch(rowbase + (size_t)(t + 5) * IN_F + fb, 0, 1);
        }

        int pk0[4], pk1[4];
        #pragma unroll
        for (int d = 0; d < 4; ++d) {
            pk0[d] = pack_bf16(hst[0][2*d], hst[0][2*d+1]);
            pk1[d] = pack_bf16(hst[1][2*d], hst[1][2*d+1]);
        }
        v8i hbw;
        #pragma unroll
        for (int d = 0; d < 4; ++d) {
            const int send = hh ? pk0[d] : pk1[d];
            const int recv = __shfl_xor(send, 16, 32);
            hbw[d]     = hh ? recv   : pk0[d];
            hbw[4 + d] = hh ? pk1[d] : recv;
        }
        const v16bf hb = as_v16bf(hbw);

        v8f zt[2], rt[2];
        #pragma unroll
        for (int mt = 0; mt < 2; ++mt) {
            v8f az = wmma_fp8(axw[0][mt], xb, zc);
            az = wmma_bf16(ahw[0][mt], hb, az);
            v8f ar = wmma_fp8(axw[1][mt], xb, zc);
            ar = wmma_bf16(ahw[1][mt], hb, ar);
            #pragma unroll
            for (int j = 0; j < 8; ++j) {
                zt[mt][j] = fast_sigmoid(az[j]);
                rt[mt][j] = fast_sigmoid(ar[j]);
            }
        }

        float rh0[8], rh1[8];
        #pragma unroll
        for (int j = 0; j < 8; ++j) {
            rh0[j] = rt[0][j] * hst[0][j];
            rh1[j] = rt[1][j] * hst[1][j];
        }
        int q0[4], q1[4];
        #pragma unroll
        for (int d = 0; d < 4; ++d) {
            q0[d] = pack_bf16(rh0[2*d], rh0[2*d+1]);
            q1[d] = pack_bf16(rh1[2*d], rh1[2*d+1]);
        }
        v8i rbw;
        #pragma unroll
        for (int d = 0; d < 4; ++d) {
            const int send = hh ? q0[d] : q1[d];
            const int recv = __shfl_xor(send, 16, 32);
            rbw[d]     = hh ? recv  : q0[d];
            rbw[4 + d] = hh ? q1[d] : recv;
        }
        const v16bf rhb = as_v16bf(rbw);

        #pragma unroll
        for (int mt = 0; mt < 2; ++mt) {
            v8f ac = wmma_fp8(axw[2][mt], xb, zc);
            ac = wmma_bf16(ahw[2][mt], rhb, ac);
            #pragma unroll
            for (int j = 0; j < 8; ++j) {
                const float htil = fast_tanh(ac[j]);
                hst[mt][j] = hst[mt][j] + zt[mt][j] * (htil - hst[mt][j]);
            }
        }
    }

    float p = 0.0f;
    #pragma unroll
    for (int mt = 0; mt < 2; ++mt)
        #pragma unroll
        for (int j = 0; j < 8; ++j)
            p += hst[mt][j] * w_out[mt * 16 + hi + j];
    p += __shfl_xor(p, 16, 32);
    if (hh == 0)
        out[b0 + n] = fast_sigmoid(p + b_out[0]);
}

extern "C" void kernel_launch(void* const* d_in, const int* in_sizes, int n_in,
                              void* d_out, int out_size, void* d_ws, size_t ws_size,
                              hipStream_t stream) {
    const float* input = (const float*)d_in[0];
    const float* w_xz  = (const float*)d_in[1];
    const float* w_hz  = (const float*)d_in[2];
    const float* b_z   = (const float*)d_in[3];
    const float* w_xr  = (const float*)d_in[4];
    const float* w_hr  = (const float*)d_in[5];
    const float* b_r   = (const float*)d_in[6];
    const float* w_xh  = (const float*)d_in[7];
    const float* w_hh  = (const float*)d_in[8];
    const float* b_h   = (const float*)d_in[9];
    const float* w_out = (const float*)d_in[10];
    const float* b_out = (const float*)d_in[11];
    float* out = (float*)d_out;

    (void)in_sizes; (void)n_in; (void)out_size; (void)d_ws; (void)ws_size;

    gru_fused_kernel<<<BATCH / 16, 32, 0, stream>>>(
        input, w_xz, w_hz, b_z, w_xr, w_hr, b_r, w_xh, w_hh, b_h,
        w_out, b_out, out);
}
